// FullAttention_635655160330
// MI455X (gfx1250) — matrix operand order repaired and checked by reading, not compiled, not run on hardware
//
#include <hip/hip_runtime.h>

typedef __attribute__((ext_vector_type(16))) _Float16 v16h;
typedef __attribute__((ext_vector_type(8)))  _Float16 v8h;
typedef __attribute__((ext_vector_type(4)))  _Float16 v4h;
typedef __attribute__((ext_vector_type(2)))  _Float16 v2h;
typedef __attribute__((ext_vector_type(8)))  float    v8f;
typedef __attribute__((ext_vector_type(4)))  float    f32x4;
typedef __attribute__((ext_vector_type(2)))  float    f32x2;
typedef __attribute__((ext_vector_type(4)))  unsigned u32x4;
typedef __attribute__((ext_vector_type(8)))  unsigned u32x8;

namespace {
constexpr int Bb = 2, Ll = 2048, Ss = 2048, Hh = 8, Ee = 64, Dd = 64;
constexpr int BLOCK_M = 64;   // query rows per workgroup (4 waves x 16 queries)
constexpr int BLOCK_N = 64;   // keys per pass / LDS tile
constexpr int KROW    = 72;   // 64 + 8 halves pad: 144B pitch, 16B-aligned rows
constexpr int BPITCH  = 68;   // bias/out tile pitch (f32) == TDM 64DW + 4DW pad

__device__ inline v8f zero8() {
  v8f z = {0.f, 0.f, 0.f, 0.f, 0.f, 0.f, 0.f, 0.f};
  return z;
}

__device__ inline v16h combine16(v8h lo, v8h hi) {
  v16h r;
#pragma unroll
  for (int i = 0; i < 8; ++i) { r[i] = lo[i]; r[8 + i] = hi[i]; }
  return r;
}

// Generic (flat) LDS pointers carry the wave-relative LDS byte offset in
// their low 32 bits.
__device__ inline unsigned lds_off(const void* p) {
  return (unsigned)(uintptr_t)p;
}
}  // namespace

// Transposed flash attention: S^T = K Q^T, softmax along the C-layout M-axis
// (register tree + one shfl_xor(16)), O^T = V^T P^T. Each wave's 16q x 64s
// f32 bias tile is fetched by ONE Tensor-Data-Mover descriptor per pass
// (tensor_load_to_lds, TENSORcnt), with TDM padding reproducing the 68-DWORD
// LDS pitch; the DMA completes underneath the QK^T WMMA chain.
__global__ __launch_bounds__(128)
void fa_fwd_wmma_t(const float* __restrict__ q, const float* __restrict__ k,
                   const float* __restrict__ v, const float* __restrict__ bias,
                   float* __restrict__ out) {
  __shared__ __align__(16) _Float16 Ks[BLOCK_N][KROW];  // [key][e]
  __shared__ __align__(16) _Float16 Vt[Dd][KROW];       // [d][key]
  __shared__ __align__(16) _Float16 Pt[4][16][KROW];    // per-wave P^T [q][key]
  __shared__ __align__(16) float Bs[4][16][BPITCH];     // per-wave bias [q][s]

  const int h    = blockIdx.x;            // head-major: 8 heads share bias lines in L2
  const int m0   = blockIdx.y * BLOCK_M;
  const int b    = blockIdx.z;

  const int tid  = threadIdx.x;
  const int wave = tid >> 5;
  const int lane = tid & 31;
  const int l16  = lane & 15;
  const int hi8  = (lane >= 16) ? 8  : 0;   // C-layout row-offset
  const int hi16 = (lane >= 16) ? 16 : 0;   // A and B K-offset (this code's contiguous k order)

  const int   R      = m0 + wave * 16;      // wave's first query row
  const int   qg     = R + l16;             // this lane's query (C-layout column)
  const float scale  = 0.125f;              // 1/sqrt(64)
  const float NEGINF = -__builtin_inff();

  // ---- TDM descriptor invariants (wave-uniform, SGPR-resident).
  // Group1: data_size=2(4B) | pad_enable | pad_interval=5(64DW) | pad_amount=3(4DW);
  //         tensor_dim0=8(H), tensor_dim1=2048(S); tile=(1,64,16);
  //         stride0=8(H), stride1=16384(S*H).
  const u32x8 g1 = {0x07520000u, 0x00080000u, 0x08000000u, 0x00010000u,
                    0x00100040u, 8u,          0x40000000u, 0u};
  const u32x4 g2 = {2048u, 0u, 0u, 0u};   // tensor_dim2 = 2048 (L)
  const u32x4 g3 = {0u, 0u, 0u, 0u};
  const unsigned bs_lds =
      __builtin_amdgcn_readfirstlane(lds_off(&Bs[wave][0][0]));

  // ---- Q^T as B operand (e x 16q): col = q = l16, k = e contiguous per lane.
  const float* qrow = q + (((size_t)b * Ll + qg) * Hh + h) * Ee;
  v16h qb[2];
#pragma unroll
  for (int c = 0; c < 2; ++c) {
    const int base = 32 * c + hi16;
    f32x4 a0 = *(const f32x4*)(qrow + base);
    f32x4 a1 = *(const f32x4*)(qrow + base + 4);
    f32x4 a2 = *(const f32x4*)(qrow + base + 8);
    f32x4 a3 = *(const f32x4*)(qrow + base + 12);
#pragma unroll
    for (int i = 0; i < 4; ++i) {
      qb[c][i]      = (_Float16)a0[i];
      qb[c][4 + i]  = (_Float16)a1[i];
      qb[c][8 + i]  = (_Float16)a2[i];
      qb[c][12 + i] = (_Float16)a3[i];
    }
  }

  // O^T accumulators: tile n covers d = n*16 + (hi8+r), col = q = l16.
  v8f o[4];
#pragma unroll
  for (int n = 0; n < 4; ++n) o[n] = zero8();
  float mcur = NEGINF, lcur = 0.f;          // per-lane scalars (one query each)

  const int nkey = m0 + BLOCK_M;            // causal bound for this block
  const float* kbase = k + (((size_t)b * Ss) * Hh + h) * Ee;
  const float* vbase = v + (((size_t)b * Ss) * Hh + h) * Dd;

  for (int j0 = 0; j0 < nkey; j0 += BLOCK_N) {
    // ---- Cooperative stage K (row-major) and V (transposed), f32 -> f16.
#pragma unroll
    for (int idx = tid; idx < (BLOCK_N * Ee) / 4; idx += 128) {  // 8 iters
      const int sl = idx >> 4;          // key row (16 float4 per row)
      const int e4 = (idx & 15) * 4;    // feature quad
      f32x4 kv = *(const f32x4*)(kbase + (size_t)(j0 + sl) * (Hh * Ee) + e4);
      f32x4 vv = *(const f32x4*)(vbase + (size_t)(j0 + sl) * (Hh * Dd) + e4);
      v4h kh;
#pragma unroll
      for (int jj = 0; jj < 4; ++jj) kh[jj] = (_Float16)kv[jj];
      *(v4h*)&Ks[sl][e4] = kh;
#pragma unroll
      for (int jj = 0; jj < 4; ++jj) Vt[e4 + jj][sl] = (_Float16)vv[jj];
    }
    __syncthreads();

    if (j0 < R + 16) {  // wave-uniform causal pass skip
      // ---- One TDM op fetches this wave's whole 16q x 64s bias tile into
      //      Bs (padded to the 68-DWORD pitch by the descriptor).
      {
        const float* bsrc =
            bias + ((size_t)(b * Ll + R) * Ss + j0) * Hh + h;
        const unsigned long long ga = (unsigned long long)(uintptr_t)bsrc;
        const unsigned galo = __builtin_amdgcn_readfirstlane((unsigned)ga);
        const unsigned gahi =
            __builtin_amdgcn_readfirstlane((unsigned)(ga >> 32));
        const u32x4 g0 = {1u,                      // count=1 valid descriptor
                          bs_lds,                  // LDS destination
                          galo,                    // global_addr[31:0]
                          (gahi & 0x01FFFFFFu) | 0x80000000u};  // addr hi | type=2
        asm volatile("tensor_load_to_lds %0, %1, %2, %3"
                     :: "s"(g0), "s"(g1), "s"(g2), "s"(g3) : "memory");
      }

      // ---- S^T = K Q^T : hoist all 16 operand b128 LDS loads, then fire
      //      the 8 WMMAs back-to-back (amortized s_wait_dscnt).
      v16h ka[4][2];
#pragma unroll
      for (int t = 0; t < 4; ++t) {
        const _Float16* kr = &Ks[t * 16 + l16][0];  // lane's key row (A operand)
#pragma unroll
        for (int c = 0; c < 2; ++c)
          ka[t][c] = combine16(*(const v8h*)(kr + 32 * c + hi16),
                               *(const v8h*)(kr + 32 * c + hi16 + 8));
      }
      v8f st[4];
#pragma unroll
      for (int t = 0; t < 4; ++t) {
        v8f acc = __builtin_amdgcn_wmma_f32_16x16x32_f16(
            false, ka[t][0], false, qb[0], (short)0, zero8(), false, false);
        st[t] = __builtin_amdgcn_wmma_f32_16x16x32_f16(
            false, ka[t][1], false, qb[1], (short)0, acc, false, false);
      }

      // ---- The bias DMA must have landed before reading Bs.
      asm volatile("s_wait_tensorcnt 0x0" ::: "memory");

      // ---- scale + bias + mask; reduction = register tree + 1 shfl_xor(16).
      float x[4][8];
      float mloc = NEGINF;
#pragma unroll
      for (int t = 0; t < 4; ++t) {
#pragma unroll
        for (int r = 0; r < 8; ++r) {
          const int sl = t * 16 + hi8 + r;            // key within tile
          float xv = st[t][r] * scale + Bs[wave][l16][sl];
          if (j0 + sl > qg) xv = NEGINF;              // causal mask
          x[t][r] = xv;
          mloc = fmaxf(mloc, xv);
        }
      }
      mloc = fmaxf(mloc, __shfl_xor(mloc, 16, 32));   // join the two key halves
      const float mnew  = fmaxf(mcur, mloc);
      const float alpha = __expf(mcur - mnew);        // exp(-inf)=0 on first pass
      float rs = 0.f;
#pragma unroll
      for (int t = 0; t < 4; ++t) {
#pragma unroll
        for (int r = 0; r < 8; r += 2) {              // packed v2h P^T stores
          const float p0 = __expf(x[t][r]     - mnew);
          const float p1 = __expf(x[t][r + 1] - mnew);
          rs += p0 + p1;
          v2h pp = {(_Float16)p0, (_Float16)p1};
          *(v2h*)&Pt[wave][l16][t * 16 + hi8 + r] = pp;
        }
      }
      rs += __shfl_xor(rs, 16, 32);
      lcur = lcur * alpha + rs;
      mcur = mnew;
#pragma unroll
      for (int n = 0; n < 4; ++n)
#pragma unroll
        for (int r = 0; r < 8; ++r) o[n][r] *= alpha;

      // ---- O^T += V^T P^T : hoist all operand loads, then 8 WMMAs.
      v16h pb[2];
#pragma unroll
      for (int c = 0; c < 2; ++c) {
        const _Float16* pr = &Pt[wave][l16][32 * c + hi16];
        pb[c] = combine16(*(const v8h*)(pr), *(const v8h*)(pr + 8));
      }
      v16h va[4][2];
#pragma unroll
      for (int n = 0; n < 4; ++n) {
        const _Float16* vr = &Vt[n * 16 + l16][0];
#pragma unroll
        for (int c = 0; c < 2; ++c)
          va[n][c] = combine16(*(const v8h*)(vr + 32 * c + hi16),
                               *(const v8h*)(vr + 32 * c + hi16 + 8));
      }
#pragma unroll
      for (int n = 0; n < 4; ++n) {
        o[n] = __builtin_amdgcn_wmma_f32_16x16x32_f16(
            false, va[n][0], false, pb[0], (short)0, o[n], false, false);
        o[n] = __builtin_amdgcn_wmma_f32_16x16x32_f16(
            false, va[n][1], false, pb[1], (short)0, o[n], false, false);
      }
    }
    __syncthreads();  // before next K/V tile overwrite
  }

  // ---- Normalize (per-lane scalar), transpose O^T -> O via per-wave LDS
  //      (float2 packed stores), then fully coalesced float4 global stores.
  const float inv = 1.0f / lcur;  // every query sees >= 1 key (diagonal)
#pragma unroll
  for (int n = 0; n < 4; ++n)
#pragma unroll
    for (int r = 0; r < 8; r += 2) {
      f32x2 val = {o[n][r] * inv, o[n][r + 1] * inv};
      *(f32x2*)&Bs[wave][l16][n * 16 + hi8 + r] = val;  // Bs[q][d]
    }

#pragma unroll
  for (int i = 0; i < 8; ++i) {
    const int flat = i * 128 + lane * 4;     // 16q x 64d, float4 per lane
    const int qq = flat >> 6, d = flat & 63;
    f32x4 val = *(const f32x4*)&Bs[wave][qq][d];
    *(f32x4*)(out + (((size_t)b * Ll + R + qq) * Hh + h) * Dd + d) = val;
  }
}

extern "C" void kernel_launch(void* const* d_in, const int* in_sizes, int n_in,
                              void* d_out, int out_size, void* d_ws, size_t ws_size,
                              hipStream_t stream) {
  (void)in_sizes; (void)n_in; (void)out_size; (void)d_ws; (void)ws_size;
  const float* q    = (const float*)d_in[0];
  const float* k    = (const float*)d_in[1];
  const float* v    = (const float*)d_in[2];
  // d_in[3] = attn_mask: pure causal triu(k=1); recomputed analytically in-kernel.
  const float* bias = (const float*)d_in[4];
  float* out = (float*)d_out;

  dim3 grid(Hh, Ll / BLOCK_M, Bb);  // head-major so 8 heads share bias cachelines
  dim3 block(128);                  // 4 waves (wave32)
  fa_fwd_wmma_t<<<grid, block, 0, stream>>>(q, k, v, bias, out);
}
